// Memo_44547400794188
// MI455X (gfx1250) — compile-verified, matrix operand order checked by reading, not run on hardware
//
#include <hip/hip_runtime.h>
#include <hip/hip_bf16.h>

typedef __attribute__((ext_vector_type(16))) _Float16 v16h;
typedef __attribute__((ext_vector_type(8)))  _Float16 v8h;
typedef __attribute__((ext_vector_type(8)))  float    v8f;

// Problem constants
#define BATCH 32
#define DIM   256
#define HW    1024        // 32*32
#define NROW  32768       // BATCH*HW
#define KCODE 1024

// ---------------------------------------------------------------------------
// CDNA5 async global->LDS copy (ASYNCcnt-tracked), 16B per lane.
// lds_off: byte offset within the workgroup LDS allocation (= low 32 bits of
// the flat shared-memory address, per ISA flat->LDS truncation).
// ---------------------------------------------------------------------------
__device__ __forceinline__ void async_copy_b128(unsigned lds_off, const void* gptr) {
    asm volatile("global_load_async_to_lds_b128 %0, %1, off"
                 :: "v"(lds_off), "v"((unsigned long long)(uintptr_t)gptr)
                 : "memory");
}
__device__ __forceinline__ void wait_asynccnt0() {
    asm volatile("s_wait_asynccnt 0x0" ::: "memory");
}

__device__ __forceinline__ v16h combine(v8h lo, v8h hi) {
    return __builtin_shufflevector(lo, hi, 0, 1, 2, 3, 4, 5, 6, 7,
                                   8, 9, 10, 11, 12, 13, 14, 15);
}

// ---------------------------------------------------------------------------
// Kernel 1: transpose z [B, D, HW] -> z32t [N, D] (fp32) and z16 [N, D] (f16)
// LDS-tiled 32x32 transpose, coalesced on both sides.
// ---------------------------------------------------------------------------
__global__ __launch_bounds__(256) void prep_z_kernel(const float* __restrict__ z,
                                                     _Float16* __restrict__ z16,
                                                     float* __restrict__ z32t) {
    __shared__ float tile[32][33];  // +1 pad: conflict-free transposed reads
    const int b   = blockIdx.z;
    const int d0  = blockIdx.y * 32;
    const int hw0 = blockIdx.x * 32;
    const int j   = threadIdx.x & 31;   // hw within tile (read phase)
    const int i   = threadIdx.x >> 5;   // 0..7

    const float* src = z + (size_t)b * (DIM * HW);
#pragma unroll
    for (int r = 0; r < 4; ++r) {
        int d = i + r * 8;
        tile[d][j] = src[(size_t)(d0 + d) * HW + hw0 + j];  // coalesced over hw
    }
    __syncthreads();

    const int ii   = threadIdx.x & 31;  // d within tile (write phase)
    const int jrow = threadIdx.x >> 5;
#pragma unroll
    for (int r = 0; r < 4; ++r) {
        int jj = jrow + r * 8;          // hw within tile
        float v = tile[ii][jj];
        size_t o = ((size_t)b * HW + hw0 + jj) * DIM + d0 + ii;  // coalesced over d
        z32t[o] = v;
        z16[o]  = (_Float16)v;
    }
}

// ---------------------------------------------------------------------------
// Kernel 2: e16[k][d] = (f16) e[k][d]; enorm[k] = ||e_k||^2. One wave per code.
// ---------------------------------------------------------------------------
__global__ __launch_bounds__(32) void prep_e_kernel(const float* __restrict__ e,
                                                    _Float16* __restrict__ e16,
                                                    float* __restrict__ enorm) {
    const int k    = blockIdx.x;
    const int lane = threadIdx.x;
    const float* row = e + (size_t)k * DIM;
    _Float16* orow   = e16 + (size_t)k * DIM;
    float s = 0.0f;
#pragma unroll
    for (int r = 0; r < 8; ++r) {
        float v = row[lane + r * 32];           // coalesced
        s += v * v;
        orow[lane + r * 32] = (_Float16)v;
    }
#pragma unroll
    for (int off = 16; off >= 1; off >>= 1) s += __shfl_xor(s, off, 32);
    if (lane == 0) enorm[k] = s;
}

// ---------------------------------------------------------------------------
// Kernel 3: WMMA distance GEMM + argmin.
//   score(n) = ||e_n||^2 - 2 * (z_m . e_n)   (||z||^2 dropped: argmin-invariant)
// Block = 8 waves; each wave owns one 16-row tile of z. The 32-code B tiles
// are shared across all 8 waves via LDS, double-buffered with CDNA5 async
// global->LDS copies. sched_group_barrier pins a software-pipelined order:
// 8 DS reads ahead, then {2 WMMA, 4 DS reads} x6, then 4 WMMAs to drain,
// so LDS latency hides behind matrix-pipe execution (partial dscnt waits).
// A-fragment per lane (16-bit A 16x32 layout): m = lane&15,
//   K halves = [kb, kb+8) and [kb+16, kb+24) with kb = 8*(lane>=16).
// ---------------------------------------------------------------------------
__global__ __launch_bounds__(256) void gemm_argmin_kernel(const _Float16* __restrict__ z16,
                                                          const _Float16* __restrict__ e16,
                                                          const float* __restrict__ enorm,
                                                          int* __restrict__ idx_out) {
    __shared__ __align__(16) _Float16 btile[2][32 * DIM];   // 2 x 16KB

    const int tid   = threadIdx.x;
    const int lane  = tid & 31;
    const int wave  = tid >> 5;
    const int mtile = blockIdx.x * 8 + wave;         // 2048 m-tiles total
    const int col   = lane & 15;
    const int kboff = ((lane >> 4) & 1) << 3;        // 0 or 8

    // Load the 8 A fragments once (resident for the whole code loop).
    v16h afrag[8];
    const _Float16* arow = z16 + (size_t)((mtile << 4) + col) * DIM;
#pragma unroll
    for (int kt = 0; kt < 8; ++kt) {
        const _Float16* p = arow + kt * 32 + kboff;
        afrag[kt] = combine(*(const v8h*)(p), *(const v8h*)(p + 16));
    }

    float best[8];
    int   bidx[8];
#pragma unroll
    for (int v = 0; v < 8; ++v) { best[v] = 3.4e38f; bidx[v] = 0; }

    const unsigned lds0 = (unsigned)(uintptr_t)(&btile[0][0]);
    const unsigned lds1 = (unsigned)(uintptr_t)(&btile[1][0]);

    // Prefetch code tile 0 (32 codes x 256 halfs = 16KB, contiguous rows).
    {
        const char* g = (const char*)e16 + tid * 16;
#pragma unroll
        for (int r = 0; r < 4; ++r)
            async_copy_b128(lds0 + tid * 16 + r * 4096, g + r * 4096);
    }
    wait_asynccnt0();
    __syncthreads();

    for (int t = 0; t < 32; ++t) {
        const int buf = t & 1;
        if (t + 1 < 32) {
            // Issue async loads for the next tile into the other buffer.
            const unsigned nxt = buf ? lds0 : lds1;
            const char* g = (const char*)e16 + (size_t)(t + 1) * (32 * DIM * 2) + tid * 16;
#pragma unroll
            for (int r = 0; r < 4; ++r)
                async_copy_b128(nxt + tid * 16 + r * 4096, g + r * 4096);
        }

        // Two 16-code sub-tiles with independent accumulators.
        const _Float16* bp0 = &btile[buf][(col) * DIM + kboff];
        const _Float16* bp1 = &btile[buf][(16 + col) * DIM + kboff];

        v8f acc0 = {};
        v8f acc1 = {};
#pragma unroll
        for (int kt = 0; kt < 8; ++kt) {
            const _Float16* q0 = bp0 + kt * 32;
            const _Float16* q1 = bp1 + kt * 32;
            v16h b0 = combine(*(const v8h*)(q0), *(const v8h*)(q0 + 16));
            v16h b1 = combine(*(const v8h*)(q1), *(const v8h*)(q1 + 16));
            acc0 = __builtin_amdgcn_wmma_f32_16x16x32_f16(
                false, afrag[kt], false, b0, (short)0, acc0, false, false);
            acc1 = __builtin_amdgcn_wmma_f32_16x16x32_f16(
                false, afrag[kt], false, b1, (short)0, acc1, false, false);
        }

        // Pin the pipelined schedule: 8 DS reads ahead, then alternate
        // {2 WMMA, 4 DS reads}, drain with 4 WMMAs. (0x100=DS read, 0x8=WMMA)
        __builtin_amdgcn_sched_group_barrier(0x100, 8, 0);
#pragma unroll
        for (int i = 0; i < 6; ++i) {
            __builtin_amdgcn_sched_group_barrier(0x008, 2, 0);
            __builtin_amdgcn_sched_group_barrier(0x100, 4, 0);
        }
        __builtin_amdgcn_sched_group_barrier(0x008, 4, 0);

        const int n0 = t * 32;
        {
            const float en   = enorm[n0 + col];
            const int   nidx = n0 + col;
#pragma unroll
            for (int v = 0; v < 8; ++v) {
                float s = en - 2.0f * acc0[v];
                if (s < best[v]) { best[v] = s; bidx[v] = nidx; }
            }
        }
        {
            const float en   = enorm[n0 + 16 + col];
            const int   nidx = n0 + 16 + col;
#pragma unroll
            for (int v = 0; v < 8; ++v) {
                float s = en - 2.0f * acc1[v];
                if (s < best[v]) { best[v] = s; bidx[v] = nidx; }
            }
        }

        wait_asynccnt0();   // my share of tile t+1 landed in LDS
        __syncthreads();    // everyone done loading t+1 and reading buf t
    }

    // Reduce argmin over the 16 lanes holding the 16 columns (stays in-half).
#pragma unroll
    for (int off = 8; off >= 1; off >>= 1) {
#pragma unroll
        for (int v = 0; v < 8; ++v) {
            float ob = __shfl_xor(best[v], off, 32);
            int   oi = __shfl_xor(bidx[v], off, 32);
            if (ob < best[v] || (ob == best[v] && oi < bidx[v])) {
                best[v] = ob; bidx[v] = oi;
            }
        }
    }
    if (col == 0) {
        // lanes 0..15 hold rows M=v; lanes 16..31 hold rows M=v+8
        int mbase = (mtile << 4) + ((lane >> 4) << 3);
#pragma unroll
        for (int v = 0; v < 8; ++v) idx_out[mbase + v] = bidx[v];
    }
}

// ---------------------------------------------------------------------------
// Kernel 4: z_q (NCHW, coalesced writes) + indices-as-float.
// ---------------------------------------------------------------------------
__global__ __launch_bounds__(256) void zq_idx_kernel(const float* __restrict__ e,
                                                     const int* __restrict__ idx,
                                                     float* __restrict__ out_zq,
                                                     float* __restrict__ out_idx) {
    const int g  = blockIdx.x * 256 + threadIdx.x;   // over B*D*HW
    const int hw = g & (HW - 1);
    const int d  = (g >> 10) & (DIM - 1);
    const int b  = g >> 18;
    const int n  = (b << 10) + hw;
    const int k  = idx[n];
    out_zq[g] = e[(size_t)k * DIM + d];              // 1MB table -> L2 resident
    if (g < NROW) out_idx[g] = (float)idx[g];
}

// ---------------------------------------------------------------------------
// Kernel 5: loss (NHWC, fully coalesced via the z32t staging buffer).
// ---------------------------------------------------------------------------
__global__ __launch_bounds__(256) void loss_kernel(const float* __restrict__ e,
                                                   const float* __restrict__ z32t,
                                                   const int* __restrict__ idx,
                                                   float* __restrict__ out_loss) {
    const int g = blockIdx.x * 256 + threadIdx.x;    // over N*D
    const int n = g >> 8;
    const int d = g & (DIM - 1);
    const int k = idx[n];
    float diff = e[(size_t)k * DIM + d] - z32t[g];
    out_loss[g] = diff * diff;
}

// ---------------------------------------------------------------------------
extern "C" void kernel_launch(void* const* d_in, const int* in_sizes, int n_in,
                              void* d_out, int out_size, void* d_ws, size_t ws_size,
                              hipStream_t stream) {
    const float* z = (const float*)d_in[0];             // [32,256,32,32]
    const float* e = (const float*)d_in[1];             // [1024,256]

    // Workspace layout (bytes)
    char* ws = (char*)d_ws;
    _Float16* z16   = (_Float16*)ws;                             // 16 MB
    _Float16* e16   = (_Float16*)(ws + (size_t)NROW * DIM * 2);  // 512 KB
    float*    enorm = (float*)((char*)e16 + (size_t)KCODE * DIM * 2);  // 4 KB
    int*      widx  = (int*)((char*)enorm + KCODE * sizeof(float));    // 128 KB
    float*    z32t  = (float*)((char*)widx + NROW * sizeof(int));      // 32 MB

    float* out_zq   = (float*)d_out;                 // 8388608 floats
    float* out_idx  = out_zq + (size_t)BATCH * DIM * HW;   // 32768 floats
    float* out_loss = out_idx + NROW;                // 8388608 floats

    // 1) transpose/convert z
    prep_z_kernel<<<dim3(HW / 32, DIM / 32, BATCH), 256, 0, stream>>>(z, z16, z32t);
    // 2) codebook convert + norms
    prep_e_kernel<<<KCODE, 32, 0, stream>>>(e, e16, enorm);
    // 3) WMMA GEMM + argmin (2048 waves, 8 waves/block)
    gemm_argmin_kernel<<<(NROW / 16) / 8, 256, 0, stream>>>(z16, e16, enorm, widx);
    // 4) z_q + indices
    zq_idx_kernel<<<(BATCH * DIM * HW) / 256, 256, 0, stream>>>(e, widx, out_zq, out_idx);
    // 5) loss
    loss_kernel<<<(NROW * DIM) / 256, 256, 0, stream>>>(e, z32t, widx, out_loss);
}
